// CRF_78391743086883
// MI455X (gfx1250) — matrix operand order repaired and checked by reading, not compiled, not run on hardware
//
#include <hip/hip_runtime.h>

typedef __attribute__((ext_vector_type(16))) _Float16 v16h;
typedef __attribute__((ext_vector_type(8)))  float    v8f;

// Problem shape (matches reference setup_inputs()).
#define CRF_B 512
#define CRF_S 1024
#define CRF_T 32

#define L2E 1.4426950408889634f   // log2(e)
#define LN2 0.6931471805599453f   // ln(2)
#define PD  4                     // register software-pipeline depth

// ---- fast cross-lane reductions (within each 16-lane half) ----------------
#define DPP8_SEL(a,b,c,d,e,f,g,h) \
    ((a)|((b)<<3)|((c)<<6)|((d)<<9)|((e)<<12)|((f)<<15)|((g)<<18)|((h)<<21))

template <int SEL>
__device__ __forceinline__ float dpp8f(float v) {
    int i = __builtin_bit_cast(int, v);
    i = __builtin_amdgcn_mov_dpp8(i, SEL);
    return __builtin_bit_cast(float, i);
}
__device__ __forceinline__ float plane16_xor8(float v) {
    int i = __builtin_bit_cast(int, v);
    // lane i (within each row of 16) reads lane i^8
    i = __builtin_amdgcn_permlane16(i, i, 0xFEDCBA98u, 0x76543210u, false, false);
    return __builtin_bit_cast(float, i);
}
__device__ __forceinline__ float rowmax32(float x0, float x1) {
    float t = fmaxf(x0, x1);
    t = fmaxf(t, dpp8f<DPP8_SEL(1,0,3,2,5,4,7,6)>(t));   // xor 1
    t = fmaxf(t, dpp8f<DPP8_SEL(2,3,0,1,6,7,4,5)>(t));   // xor 2
    t = fmaxf(t, dpp8f<DPP8_SEL(4,5,6,7,0,1,2,3)>(t));   // xor 4
    t = fmaxf(t, plane16_xor8(t));                        // xor 8
    return t;
}
__device__ __forceinline__ float rowsum16(float t) {
    t += dpp8f<DPP8_SEL(1,0,3,2,5,4,7,6)>(t);
    t += dpp8f<DPP8_SEL(2,3,0,1,6,7,4,5)>(t);
    t += dpp8f<DPP8_SEL(4,5,6,7,0,1,2,3)>(t);
    t += plane16_xor8(t);
    return t;
}

// ---------------------------------------------------------------------------
// Partition (forward algorithm). One wave (32 lanes) owns 16 batches.
// alpha[16 batches][32 tags] = two 16x16 f32 WMMA C/D tiles:
//   reg r, lane L: b_local = r + 8*(L>>4), tag j = (L&15) + 16*tile.
// Step: DPP rowmax -> p=exp(alpha-M) -> LDS transpose to A layout ->
//       2x v_wmma_f32_16x16x32_f16 vs constant E=exp(trans) ->
//       alpha' = M + e + ln(acc), masked blend.
// Emissions/mask come from a PD-deep register ring filled PD steps ahead;
// WGP-scope global_prefetch_b8 runs 16 steps ahead of that.
// ---------------------------------------------------------------------------
__global__ __launch_bounds__(32, 1) void crf_partition_kernel(
    const float* __restrict__ em, const float* __restrict__ mask,
    const float* __restrict__ trans, const float* __restrict__ start_t,
    const float* __restrict__ end_t, float* __restrict__ part_out)
{
    const int lane = threadIdx.x;        // 0..31
    const int g    = blockIdx.x;         // batch group of 16
    const int half = lane >> 4;          // 0 or 1
    const int l15  = lane & 15;

    __shared__ float p_lds[16][CRF_T + 1];

    // ---- constant B tiles: E[k][j] = exp(trans[k][j]) ----
    // B layout (32x16 f16): lane -> column j = l15 (+16*tile);
    // element e of v16h -> k = 16*half + e.
    v16h b0v, b1v;
    #pragma unroll
    for (int e = 0; e < 16; ++e) {
        int k = half * 16 + e;
        b0v[e] = (_Float16)__builtin_amdgcn_exp2f(trans[k * CRF_T + l15] * L2E);
        b1v[e] = (_Float16)__builtin_amdgcn_exp2f(trans[k * CRF_T + 16 + l15] * L2E);
    }

    // ---- per-row global bases ----
    size_t ebase[8], mbase[8];
    #pragma unroll
    for (int r = 0; r < 8; ++r) {
        int bl = r + 8 * half;
        ebase[r] = (size_t)(g * 16 + bl) * CRF_S * CRF_T + l15;
        mbase[r] = (size_t)(g * 16 + bl) * CRF_S;
    }
    // prefetch base: lane covers (row = l15, 64B half-chunk = half)
    const size_t pfbase = (size_t)(g * 16 + l15) * CRF_S * CRF_T + half * 16;

    // ---- alpha init (step 0) ----
    float a0[8], a1[8];
    #pragma unroll
    for (int r = 0; r < 8; ++r) {
        a0[r] = start_t[l15]      + em[ebase[r]];        // em[b, 0, l15]
        a1[r] = start_t[16 + l15] + em[ebase[r] + 16];   // em[b, 0, 16+l15]
    }

    // ---- register pipeline ring ----
    float pe0[PD][8], pe1[PD][8], pm[PD][8];

#define LOAD_SLOT(s_, slot_)                                                  \
    do {                                                                      \
        _Pragma("unroll")                                                     \
        for (int r = 0; r < 8; ++r) {                                         \
            size_t eb_ = ebase[r] + (size_t)(s_) * CRF_T;                     \
            pe0[slot_][r] = em[eb_];                                          \
            pe1[slot_][r] = em[eb_ + 16];                                     \
            pm[slot_][r]  = mask[mbase[r] + (s_)];                            \
        }                                                                     \
    } while (0)

#define DO_STEP(s_, slot_)                                                    \
    do {                                                                      \
        if ((s_) + 16 < CRF_S)  /* WGP-scope prefetch: fill all cache levels */ \
            __builtin_prefetch(em + pfbase + (size_t)((s_) + 16) * CRF_T, 0, 3); \
        float M[8];                                                           \
        _Pragma("unroll")                                                     \
        for (int r = 0; r < 8; ++r) M[r] = rowmax32(a0[r], a1[r]);            \
        _Pragma("unroll")                                                     \
        for (int r = 0; r < 8; ++r) {                                         \
            int bl_ = r + 8 * half;                                           \
            p_lds[bl_][l15]      = __builtin_amdgcn_exp2f((a0[r] - M[r]) * L2E); \
            p_lds[bl_][16 + l15] = __builtin_amdgcn_exp2f((a1[r] - M[r]) * L2E); \
        }                                                                     \
        __syncthreads();                                                      \
        v16h a_;                                                              \
        {                                                                     \
            int k0_ = half * 16;                                              \
            _Pragma("unroll")                                                 \
            for (int e = 0; e < 8; ++e) {                                     \
                a_[e]     = (_Float16)p_lds[l15][k0_ + e];                    \
                a_[8 + e] = (_Float16)p_lds[l15][k0_ + 8 + e];                \
            }                                                                 \
        }                                                                     \
        __syncthreads();                                                      \
        const v8f cz_ = {};                                                   \
        v8f acc0_ = __builtin_amdgcn_wmma_f32_16x16x32_f16(                   \
            false, a_, false, b0v, (short)0, cz_, false, false);              \
        v8f acc1_ = __builtin_amdgcn_wmma_f32_16x16x32_f16(                   \
            false, a_, false, b1v, (short)0, cz_, false, false);              \
        _Pragma("unroll")                                                     \
        for (int r = 0; r < 8; ++r) {                                         \
            float m_  = pm[slot_][r];                                         \
            float n0_ = M[r] + pe0[slot_][r] + LN2 * __builtin_amdgcn_logf(acc0_[r]); \
            float n1_ = M[r] + pe1[slot_][r] + LN2 * __builtin_amdgcn_logf(acc1_[r]); \
            a0[r] = n0_ * m_ + a0[r] * (1.0f - m_);                           \
            a1[r] = n1_ * m_ + a1[r] * (1.0f - m_);                           \
        }                                                                     \
        if ((s_) + PD <= CRF_S - 1) LOAD_SLOT((s_) + PD, slot_);              \
    } while (0)

    // prologue: fill the ring for s = 1..4
    LOAD_SLOT(1, 0); LOAD_SLOT(2, 1); LOAD_SLOT(3, 2); LOAD_SLOT(4, 3);

    // main loop: s = 1..1020 (slot invariant: slot(s) = (s-1)&3)
    for (int sb = 1; sb + 3 <= CRF_S - 4; sb += 4) {
        DO_STEP(sb + 0, 0);
        DO_STEP(sb + 1, 1);
        DO_STEP(sb + 2, 2);
        DO_STEP(sb + 3, 3);
    }
    // tail: s = 1021, 1022, 1023
    DO_STEP(CRF_S - 3, 0);
    DO_STEP(CRF_S - 2, 1);
    DO_STEP(CRF_S - 1, 2);

#undef DO_STEP
#undef LOAD_SLOT

    // ---- finalize: + end_transitions, logsumexp over tags ----
    #pragma unroll
    for (int r = 0; r < 8; ++r) {
        float x0 = a0[r] + end_t[l15];
        float x1 = a1[r] + end_t[16 + l15];
        float t  = rowmax32(x0, x1);
        float ss = __builtin_amdgcn_exp2f((x0 - t) * L2E)
                 + __builtin_amdgcn_exp2f((x1 - t) * L2E);
        ss = rowsum16(ss);
        if (l15 == 0)
            part_out[g * 16 + r + 8 * half] = t + LN2 * __builtin_amdgcn_logf(ss);
    }
}

// ---------------------------------------------------------------------------
// Gold-path score: one block per batch, deterministic tree reduction.
// ---------------------------------------------------------------------------
__global__ __launch_bounds__(256) void crf_score_kernel(
    const float* __restrict__ em, const long long* __restrict__ tags,
    const float* __restrict__ mask, const float* __restrict__ trans,
    const float* __restrict__ start_t, const float* __restrict__ end_t,
    float* __restrict__ score_out)
{
    const int b   = blockIdx.x;
    const int tid = threadIdx.x;
    __shared__ float red[256];
    __shared__ float redm[256];

    const long long* tb = tags + (size_t)b * CRF_S;
    const float*     mb = mask + (size_t)b * CRF_S;
    const float*     eb = em   + (size_t)b * CRF_S * CRF_T;

    float acc = 0.0f, msum = 0.0f;
    for (int s = tid; s < CRF_S; s += 256) {
        float m = mb[s];
        msum += m;
        if (s >= 1) {
            int tc = (int)tb[s];
            int tp = (int)tb[s - 1];
            acc += (eb[s * CRF_T + tc] + trans[tp * CRF_T + tc]) * m;
        }
    }
    red[tid] = acc; redm[tid] = msum;
    __syncthreads();
    for (int off = 128; off > 0; off >>= 1) {
        if (tid < off) { red[tid] += red[tid + off]; redm[tid] += redm[tid + off]; }
        __syncthreads();
    }
    if (tid == 0) {
        int t0 = (int)tb[0];
        float sc = start_t[t0] + eb[t0] + red[0];
        int last = (int)(redm[0] + 0.5f) - 1;
        if (last < 0) last = 0;
        if (last >= CRF_S) last = CRF_S - 1;
        sc += end_t[(int)tb[last]];
        score_out[b] = sc;
    }
}

// ---------------------------------------------------------------------------
// Final: out = sum_b (partition[b] - score[b])  == -(score - partition).sum()
// ---------------------------------------------------------------------------
__global__ __launch_bounds__(256) void crf_reduce_kernel(
    const float* __restrict__ part, const float* __restrict__ score,
    float* __restrict__ out)
{
    __shared__ float red[256];
    const int tid = threadIdx.x;
    float acc = 0.0f;
    for (int b = tid; b < CRF_B; b += 256) acc += part[b] - score[b];
    red[tid] = acc;
    __syncthreads();
    for (int off = 128; off > 0; off >>= 1) {
        if (tid < off) red[tid] += red[tid + off];
        __syncthreads();
    }
    if (tid == 0) out[0] = red[0];
}

extern "C" void kernel_launch(void* const* d_in, const int* in_sizes, int n_in,
                              void* d_out, int out_size, void* d_ws, size_t ws_size,
                              hipStream_t stream) {
    const float*     em      = (const float*)d_in[0];      // (B,S,T) f32
    const long long* tags    = (const long long*)d_in[1];  // (B,S) i64
    const float*     mask    = (const float*)d_in[2];      // (B,S) f32
    const float*     trans   = (const float*)d_in[3];      // (T,T) f32
    const float*     start_t = (const float*)d_in[4];      // (T,) f32
    const float*     end_t   = (const float*)d_in[5];      // (T,) f32
    float*           out     = (float*)d_out;

    float* score_ws = (float*)d_ws;          // CRF_B floats
    float* part_ws  = score_ws + CRF_B;      // CRF_B floats

    crf_score_kernel<<<CRF_B, 256, 0, stream>>>(em, tags, mask, trans,
                                                start_t, end_t, score_ws);
    crf_partition_kernel<<<CRF_B / 16, 32, 0, stream>>>(em, mask, trans,
                                                        start_t, end_t, part_ws);
    crf_reduce_kernel<<<1, 256, 0, stream>>>(part_ws, score_ws, out);
}
